// _PromptPool_944892805307
// MI455X (gfx1250) — matrix operand order repaired and checked by reading, not compiled, not run on hardware
//
#include <hip/hip_runtime.h>
#include <math.h>

typedef __bf16  v16bf __attribute__((ext_vector_type(16)));
typedef __bf16  v8bf  __attribute__((ext_vector_type(8)));
typedef float   v8f   __attribute__((ext_vector_type(8)));

#define B_ 4096
#define P_ 1024
#define L_ 8
#define D_ 768
#define K_ 5

// float -> bf16 with round-to-nearest-even, stored as raw u16
__device__ __forceinline__ unsigned short f2bf(float f) {
    unsigned int u = __float_as_uint(f);
    u += 0x7FFFu + ((u >> 16) & 1u);
    return (unsigned short)(u >> 16);
}

// ---------------- row L2-normalize, f32 -> bf16 ----------------
__global__ void normalize_rows_kernel(const float* __restrict__ src,
                                      unsigned short* __restrict__ dst) {
    const int row = blockIdx.x;
    const int t   = threadIdx.x;
    const float* s = src + (size_t)row * D_;
    __shared__ float red[256];
    float ssq = 0.f;
    for (int j = t; j < D_; j += 256) { float v = s[j]; ssq += v * v; }
    red[t] = ssq; __syncthreads();
    for (int st = 128; st > 0; st >>= 1) {
        if (t < st) red[t] += red[t + st];
        __syncthreads();
    }
    const float inv = 1.0f / fmaxf(sqrtf(red[0]), 1e-8f);
    unsigned short* d = dst + (size_t)row * D_;
    for (int j = t; j < D_; j += 256) d[j] = f2bf(s[j] * inv);
}

// ---------------- init counts + accumulator ----------------
__global__ void init_kernel(int* __restrict__ counts, double* __restrict__ accum) {
    const int t = blockIdx.x * blockDim.x + threadIdx.x;
    if (t < P_) counts[t] = 0;
    if (t == 0) *accum = 0.0;
}

// ---------------- WMMA bf16 GEMM: dist = 1 - qn * kn^T ----------------
// one wave per 16x16 tile of (B_ x P_); 4 waves per block
//
// A fragment (16x32 bf16, M x K), per ISA dense-A table:
//   lane<16 : M=lane,    elems [0..7]=K kb..kb+7, [8..15]=K kb+16..kb+23, kb=k0
//   lane>=16: M=lane-16, same with kb=k0+8
// B fragment (32x16 bf16, K x N), same k order as A:
//   lane<16 : N=lane,    elems [0..7]=K kb..kb+7, [8..15]=K kb+16..kb+23, kb=k0
//   lane>=16: N=lane-16, same with kb=k0+8
// B[k][n] = kn[n][k] (kn row-major P x D), so each lane reads two 16-byte chunks.
__global__ void gemm_dist_kernel(const unsigned short* __restrict__ qn,
                                 const unsigned short* __restrict__ kn,
                                 float* __restrict__ dist) {
    const int lane = threadIdx.x & 31;
    const int wave = threadIdx.x >> 5;
    const int tile = blockIdx.x * 4 + wave;
    const int TN   = P_ / 16;               // 64 tiles along N
    const int tm   = tile / TN;
    const int tn   = tile % TN;

    const int rowA  = tm * 16 + (lane & 15); // query row for A fragment
    const int rowB  = tn * 16 + (lane & 15); // key row (column of B) for B fragment
    const int koffA = (lane & 16) ? 8 : 0;
    const int koffB = (lane & 16) ? 8 : 0;

    const unsigned short* pa = qn + (size_t)rowA * D_;
    const unsigned short* pb = kn + (size_t)rowB * D_;

    v8f c = {};
    for (int k0 = 0; k0 < D_; k0 += 32) {
        const int ka = k0 + koffA;
        const int kb = k0 + koffB;
        v8bf alo = *(const v8bf*)(pa + ka);
        v8bf ahi = *(const v8bf*)(pa + ka + 16);
        v8bf blo = *(const v8bf*)(pb + kb);
        v8bf bhi = *(const v8bf*)(pb + kb + 16);
        v16bf a = __builtin_shufflevector(alo, ahi,
                    0,1,2,3,4,5,6,7,8,9,10,11,12,13,14,15);
        v16bf b = __builtin_shufflevector(blo, bhi,
                    0,1,2,3,4,5,6,7,8,9,10,11,12,13,14,15);
        c = __builtin_amdgcn_wmma_f32_16x16x32_bf16(
                false, a, false, b, (short)0, c, false, false);
    }

    const int n     = tn * 16 + (lane & 15);
    const int mbase = tm * 16 + ((lane & 16) ? 8 : 0);
#pragma unroll
    for (int i = 0; i < 8; ++i)
        dist[(size_t)(mbase + i) * P_ + n] = 1.0f - c[i];
}

// ---------------- top-5 smallest per row + histogram ----------------
__global__ void topk_kernel(const float* __restrict__ dist,
                            int* __restrict__ idx,
                            int* __restrict__ counts) {
    const int b = blockIdx.x;
    const int t = threadIdx.x;
    const float* row = dist + (size_t)b * P_;
    __shared__ float sv[256];
    __shared__ int   si[256];
    __shared__ int   chosen[K_];

    for (int k = 0; k < K_; ++k) {
        float best = INFINITY;
        int   bi   = 0x7fffffff;
        for (int p = t; p < P_; p += 256) {
            bool skip = false;
            for (int j = 0; j < k; ++j) if (chosen[j] == p) skip = true;
            const float v = row[p];
            if (!skip && (v < best || (v == best && p < bi))) { best = v; bi = p; }
        }
        sv[t] = best; si[t] = bi;
        __syncthreads();
        for (int s = 128; s > 0; s >>= 1) {
            if (t < s) {
                const float v2 = sv[t + s]; const int i2 = si[t + s];
                if (v2 < sv[t] || (v2 == sv[t] && i2 < si[t])) { sv[t] = v2; si[t] = i2; }
            }
            __syncthreads();
        }
        if (t == 0) {
            chosen[k] = si[0];
            idx[b * K_ + k] = si[0];
            atomicAdd(&counts[si[0]], 1);
        }
        __syncthreads();
    }
}

// ---------------- gather prompts[idx] -> selected (pure b128 streaming) ----
// one block per (b,k,l) row of 768 floats; 192 threads x float4
__global__ void gather_kernel(const float* __restrict__ prompts,
                              const int* __restrict__ idx,
                              float* __restrict__ out) {
    const int row = blockIdx.x;            // b*K*L + k*L + l
    const int l   = row & (L_ - 1);
    const int bk  = row >> 3;              // b*K + k  (L_=8)
    const int p   = idx[bk];
    const float4* src = (const float4*)(prompts + ((size_t)p * L_ + l) * D_);
    float4*       dst = (float4*)(out + (size_t)row * D_);
    dst[threadIdx.x] = src[threadIdx.x];
}

// ---------------- count-weighted sum of dist ----------------
__global__ void wsum_kernel(const float* __restrict__ dist,
                            const int* __restrict__ counts,
                            double* __restrict__ accum) {
    const int t = threadIdx.x;
    const size_t n = (size_t)B_ * P_;
    float local = 0.f;
    for (size_t i = (size_t)blockIdx.x * blockDim.x + t; i < n;
         i += (size_t)gridDim.x * blockDim.x)
        local += dist[i] * (float)counts[i & (P_ - 1)];
    __shared__ float red[256];
    red[t] = local; __syncthreads();
    for (int s = 128; s > 0; s >>= 1) {
        if (t < s) red[t] += red[t + s];
        __syncthreads();
    }
    if (t == 0) atomicAdd(accum, (double)red[0]);
}

__global__ void finalize_kernel(const double* __restrict__ accum,
                                float* __restrict__ out_mean) {
    if (threadIdx.x == 0)
        *out_mean = (float)(*accum / ((double)B_ * (double)B_ * (double)K_));
}

extern "C" void kernel_launch(void* const* d_in, const int* in_sizes, int n_in,
                              void* d_out, int out_size, void* d_ws, size_t ws_size,
                              hipStream_t stream) {
    const float* query   = (const float*)d_in[0];   // (B, D)
    const float* keys    = (const float*)d_in[1];   // (P, D)
    const float* prompts = (const float*)d_in[2];   // (P, L, D)
    float* out = (float*)d_out;                     // selected (B*K*L*D) + mean (1)

    char* ws = (char*)d_ws;
    unsigned short* qn     = (unsigned short*)(ws);                         // B*D bf16
    unsigned short* kn     = (unsigned short*)(ws + 6291456);               // P*D bf16
    float*          dist   = (float*)(ws + 7864320);                        // B*P f32
    int*            idx    = (int*)(ws + 24641536);                         // B*K
    int*            counts = (int*)(ws + 24723456);                         // P
    double*         accum  = (double*)(ws + 24727552);                      // 1

    normalize_rows_kernel<<<P_, 256, 0, stream>>>(keys, kn);
    normalize_rows_kernel<<<B_, 256, 0, stream>>>(query, qn);
    init_kernel<<<4, 256, 0, stream>>>(counts, accum);

    gemm_dist_kernel<<<(B_ / 16) * (P_ / 16) / 4, 128, 0, stream>>>(qn, kn, dist);

    topk_kernel<<<B_, 256, 0, stream>>>(dist, idx, counts);

    gather_kernel<<<B_ * K_ * L_, 192, 0, stream>>>(prompts, idx, out);

    wsum_kernel<<<4096, 256, 0, stream>>>(dist, counts, accum);
    finalize_kernel<<<1, 32, 0, stream>>>(accum, out + (size_t)B_ * K_ * L_ * D_);
}
